// HAN_1228360647044
// MI455X (gfx1250) — matrix operand order repaired and checked by reading, not compiled, not run on hardware
//
#include <hip/hip_runtime.h>
#include <hip/hip_bf16.h>
#include <limits.h>

// ---------------- problem constants ----------------
constexpr int NN   = 50000;    // nodes
constexpr int EE   = 1000000;  // edges per metapath
constexpr int FIN  = 512;
constexpr int HID  = 128;
constexpr int HEADS= 8;
constexpr int DH   = 16;       // HID/HEADS
constexpr float NEG_SLOPE = 0.2f;

typedef __attribute__((ext_vector_type(16))) _Float16 v16h;
typedef __attribute__((ext_vector_type(8)))  float    v8f;

union AFrag { v16h v; uint4 q[2]; };
union H8    { _Float16 h[8]; uint4 q; };

// monotonic int encoding of float for atomicMax
__device__ __forceinline__ int   enc_f(float f){ int i=__float_as_int(f); return i>=0 ? i : (i^0x7fffffff); }
__device__ __forceinline__ float dec_f(int i){ return __int_as_float(i>=0 ? i : (i^0x7fffffff)); }
__device__ __forceinline__ float leaky(float a){ return a > 0.f ? a : NEG_SLOPE*a; }
__device__ __forceinline__ void load8f(const float* p, float* v){
  float4 a = ((const float4*)p)[0], b = ((const float4*)p)[1];
  v[0]=a.x; v[1]=a.y; v[2]=a.z; v[3]=a.w; v[4]=b.x; v[5]=b.y; v[6]=b.z; v[7]=b.w;
}
__device__ __forceinline__ void atomAddF(float* p, float v){ unsafeAtomicAdd(p, v); }

// ---------------- init workspace ----------------
__global__ void init_ws(float* __restrict__ o, float* __restrict__ denom,
                        int* __restrict__ amax, float* __restrict__ colsum) {
  size_t i = blockIdx.x*(size_t)blockDim.x + threadIdx.x;
  size_t st = (size_t)gridDim.x*blockDim.x;
  for (size_t j=i; j < (size_t)2*NN*HID; j+=st) o[j] = 0.f;
  for (size_t j=i; j < (size_t)2*NN*HEADS; j+=st) { denom[j]=0.f; amax[j]=INT_MIN; }
  for (size_t j=i; j < 256; j+=st) colsum[j]=0.f;
}

// ---------------- convert x fp32 -> f16 ----------------
__global__ void cvt_x(const float* __restrict__ x, _Float16* __restrict__ xh) {
  size_t i = (blockIdx.x*(size_t)blockDim.x + threadIdx.x) * 8;
  if (i >= (size_t)NN*FIN) return;
  float4 f0 = *(const float4*)(x+i);
  float4 f1 = *(const float4*)(x+i+4);
  H8 u;
  u.h[0]=(_Float16)f0.x; u.h[1]=(_Float16)f0.y; u.h[2]=(_Float16)f0.z; u.h[3]=(_Float16)f0.w;
  u.h[4]=(_Float16)f1.x; u.h[5]=(_Float16)f1.y; u.h[6]=(_Float16)f1.z; u.h[7]=(_Float16)f1.w;
  *(uint4*)(xh+i) = u.q;
}

// ---------------- pre-swizzle W_proj / Wk into WMMA B-fragment order ----------
// B 32x16 f16 order this code uses (wave32, contiguous k): lane n<16 holds col n, K = 0..15 (2 per VGPR);
// lane 16+n holds col n, K = 16..31.  Fragment = 32 lanes x 16 halves contiguous.
__global__ void swizzle_w(const float* __restrict__ Wproj, const float* __restrict__ Wk,
                          _Float16* __restrict__ wpf, _Float16* __restrict__ wkf) {
  int idx = blockIdx.x*blockDim.x + threadIdx.x;
  if (idx < 8*16*32*16) { // W_proj: 8 n_tiles, 16 k_steps
    int i    =  idx        & 15;
    int lane = (idx >> 4)  & 31;
    int ks   = (idx >> 9)  & 15;
    int nt   =  idx >> 13;
    int k    = ks*32 + (lane>>4)*16 + i;
    int col  = nt*16 + (lane&15);
    wpf[idx] = (_Float16)Wproj[k*HID + col];
  } else {
    int t = idx - 8*16*32*16;
    if (t >= 8*4*32*16) return; // Wk: 8 n_tiles, 4 k_steps
    int i    =  t        & 15;
    int lane = (t >> 4)  & 31;
    int ks   = (t >> 9)  & 3;
    int nt   =  t >> 11;
    int k    = ks*32 + (lane>>4)*16 + i;
    int col  = nt*16 + (lane&15);
    wkf[t] = (_Float16)Wk[k*HID + col];
  }
}

// ---------------- projection GEMM: xp = x @ W_proj + b  (WMMA f16->f32) -------
// block = 256 (8 waves); block owns one 16-row M tile; wave w owns N tile w.
__global__ __launch_bounds__(256) void gemm_proj(const _Float16* __restrict__ xh,
                                                 const _Float16* __restrict__ wpf,
                                                 const float* __restrict__ bias,
                                                 float* __restrict__ xp) {
  __shared__ _Float16 sA[16*520];           // 16 rows x 512, +8 pad vs LDS banks
  const int mt  = blockIdx.x;               // 0..3124
  const int row0= mt*16;
  const int tid = threadIdx.x;
  { // stage A tile: each thread loads 32 halves (4 x 16B)
    int r = tid >> 4, c = (tid & 15) * 32;
    const _Float16* g = xh + (size_t)(row0 + r)*FIN + c;
    _Float16* l = sA + r*520 + c;
    #pragma unroll
    for (int j = 0; j < 4; ++j) *(uint4*)(l + j*8) = *(const uint4*)(g + j*8);
  }
  __syncthreads();
  const int wave = tid >> 5, lane = tid & 31;
  const int arow = lane & 15, asel = lane >> 4;
  v8f acc = {};
  #pragma unroll
  for (int ks = 0; ks < 16; ++ks) {
    AFrag a, b;
    const _Float16* ap = sA + arow*520 + ks*32 + asel*16;    // A 16x32 f16 layout
    a.q[0] = *(const uint4*)(ap);
    a.q[1] = *(const uint4*)(ap + 8);
    const _Float16* bp = wpf + ((size_t)(wave*16 + ks))*512 + lane*16;
    b.q[0] = *(const uint4*)(bp);
    b.q[1] = *(const uint4*)(bp + 8);
    acc = __builtin_amdgcn_wmma_f32_16x16x32_f16(false, a.v, false, b.v,
                                                 (short)0, acc, false, false);
  }
  const int col  = wave*16 + (lane & 15);
  const float bc = bias[col];
  const int rb   = row0 + asel*8;
  #pragma unroll
  for (int v = 0; v < 8; ++v)
    xp[(size_t)(rb + v)*HID + col] = acc[v] + bc;
}

// ---------------- per-node attention scores for both metapaths ---------------
__global__ void node_scores(const float* __restrict__ xp,
                            const float* __restrict__ as0w, const float* __restrict__ ad0w,
                            const float* __restrict__ as1w, const float* __restrict__ ad1w,
                            float* __restrict__ as0, float* __restrict__ ad0,
                            float* __restrict__ as1, float* __restrict__ ad1) {
  int i = blockIdx.x*blockDim.x + threadIdx.x;       // n*8 + h
  if (i >= NN*HEADS) return;
  int h = i & 7, n = i >> 3;
  const float* xr = xp + (size_t)n*HID + h*DH;
  float s0=0,d0=0,s1=0,d1=0;
  #pragma unroll
  for (int k=0;k<DH;k++){ float v=xr[k];
    s0+=v*as0w[h*DH+k]; d0+=v*ad0w[h*DH+k]; s1+=v*as1w[h*DH+k]; d1+=v*ad1w[h*DH+k]; }
  as0[i]=s0; ad0[i]=d0; as1[i]=s1; ad1[i]=d1;
}

// ---------------- edge phase 1: segment max via monotonic-int atomicMax ------
__global__ void edge_max(const int* __restrict__ ei, const float* __restrict__ as,
                         const float* __restrict__ ad, int* __restrict__ amax) {
  int e = blockIdx.x*blockDim.x + threadIdx.x;
  if (e >= EE) return;
  int s = ei[e], d = ei[EE + e];
  float vs[8], vd[8];
  load8f(as + (size_t)s*8, vs); load8f(ad + (size_t)d*8, vd);
  #pragma unroll
  for (int h=0;h<8;h++) atomicMax(&amax[(size_t)d*8+h], enc_f(leaky(vs[h]+vd[h])));
}

// ---------------- edge phase 2: denom = segment_sum(exp(alpha - max)) --------
__global__ void edge_den(const int* __restrict__ ei, const float* __restrict__ as,
                         const float* __restrict__ ad, const int* __restrict__ amax,
                         float* __restrict__ denom) {
  int e = blockIdx.x*blockDim.x + threadIdx.x;
  if (e >= EE) return;
  int s = ei[e], d = ei[EE + e];
  float vs[8], vd[8];
  load8f(as + (size_t)s*8, vs); load8f(ad + (size_t)d*8, vd);
  #pragma unroll
  for (int h=0;h<8;h++) {
    float mx = dec_f(amax[(size_t)d*8+h]); if (!isfinite(mx)) mx = 0.f;
    atomAddF(&denom[(size_t)d*8+h], __expf(leaky(vs[h]+vd[h]) - mx));
  }
}

// ---------------- edge phase 3: scatter weighted messages (wave per edge) ----
__global__ __launch_bounds__(256) void edge_msg(const int* __restrict__ ei,
                         const float* __restrict__ as, const float* __restrict__ ad,
                         const int* __restrict__ amax, const float* __restrict__ denom,
                         const float* __restrict__ xp, float* __restrict__ o) {
  int lane = threadIdx.x & 31;
  int e = blockIdx.x*(blockDim.x>>5) + (threadIdx.x>>5);
  if (e >= EE) return;
  int s = ei[e], d = ei[EE + e];
  int h = lane >> 2;                                   // lane covers dims lane*4..+3
  float a  = leaky(as[(size_t)s*8+h] + ad[(size_t)d*8+h]);
  float mx = dec_f(amax[(size_t)d*8+h]); if (!isfinite(mx)) mx = 0.f;
  float w  = __expf(a - mx) / (denom[(size_t)d*8+h] + 1e-16f);
  float4 xv = *(const float4*)(xp + (size_t)s*HID + lane*4);   // coalesced 512B/edge
  float* op = o + (size_t)d*HID + lane*4;
  atomAddF(op+0, w*xv.x); atomAddF(op+1, w*xv.y);
  atomAddF(op+2, w*xv.z); atomAddF(op+3, w*xv.w);
}

// ---------------- relu + f16 copy of metapath outputs ------------------------
__global__ void relu_cvt(float* __restrict__ o, _Float16* __restrict__ o16) {
  size_t st = (size_t)gridDim.x*blockDim.x;
  for (size_t i = blockIdx.x*(size_t)blockDim.x + threadIdx.x;
       i < (size_t)2*NN*HID; i += st) {
    float v = fmaxf(o[i], 0.f);
    o[i] = v; o16[i] = (_Float16)v;
  }
}

// ---------------- semantic GEMM: colsum_m[c] = sum_n tanh(o_m@Wk + bk)[n,c] --
__global__ __launch_bounds__(256) void gemm_sem(const _Float16* __restrict__ o16,
                                                const _Float16* __restrict__ wkf,
                                                const float* __restrict__ bk,
                                                float* __restrict__ colsum) {
  __shared__ _Float16 sA[16*136];
  __shared__ float colred[HID];
  const int mt = blockIdx.x, m = blockIdx.y;
  const int row0 = mt*16, tid = threadIdx.x;
  if (tid < HID) colred[tid] = 0.f;
  { // stage 16x128 f16 tile
    int r = tid >> 4, c = (tid & 15) * 8;
    *(uint4*)(sA + r*136 + c) =
      *(const uint4*)(o16 + (size_t)m*NN*HID + (size_t)(row0 + r)*HID + c);
  }
  __syncthreads();
  const int wave = tid >> 5, lane = tid & 31;
  const int arow = lane & 15, asel = lane >> 4;
  v8f acc = {};
  #pragma unroll
  for (int ks = 0; ks < 4; ++ks) {
    AFrag a, b;
    const _Float16* ap = sA + arow*136 + ks*32 + asel*16;
    a.q[0] = *(const uint4*)(ap);
    a.q[1] = *(const uint4*)(ap + 8);
    const _Float16* bp = wkf + ((size_t)(wave*4 + ks))*512 + lane*16;
    b.q[0] = *(const uint4*)(bp);
    b.q[1] = *(const uint4*)(bp + 8);
    acc = __builtin_amdgcn_wmma_f32_16x16x32_f16(false, a.v, false, b.v,
                                                 (short)0, acc, false, false);
  }
  const int col = wave*16 + (lane & 15);
  const float bc = bk[col];
  float part = 0.f;
  #pragma unroll
  for (int v = 0; v < 8; ++v) part += tanhf(acc[v] + bc);
  atomicAdd(&colred[col], part);            // ds_add_f32
  __syncthreads();
  if (tid < HID) atomAddF(&colsum[m*HID + tid], colred[tid]);
}

// ---------------- semantic softmax beta --------------------------------------
__global__ __launch_bounds__(128) void sem_beta(const float* __restrict__ colsum,
                                                const float* __restrict__ q,
                                                float* __restrict__ beta) {
  __shared__ float r0[HID], r1[HID];
  int h = threadIdx.x;
  r0[h] = colsum[h]      * q[h];
  r1[h] = colsum[HID+h]  * q[h];
  __syncthreads();
  for (int s = 64; s > 0; s >>= 1) {
    if (h < s) { r0[h] += r0[h+s]; r1[h] += r1[h+s]; }
    __syncthreads();
  }
  if (h == 0) {
    float s0 = r0[0]/(float)NN, s1 = r1[0]/(float)NN;
    float mx = fmaxf(s0, s1);
    float e0 = __expf(s0-mx), e1 = __expf(s1-mx);
    beta[0] = e0/(e0+e1); beta[1] = e1/(e0+e1);
  }
}

// ---------------- fused output + final linear (128 -> 3) ---------------------
__global__ __launch_bounds__(128) void final_lin(const float* __restrict__ o,
                                                 const float* __restrict__ beta,
                                                 const float* __restrict__ Wl,
                                                 const float* __restrict__ bl,
                                                 float* __restrict__ out) {
  __shared__ float red[3][HID];
  int n = blockIdx.x, h = threadIdx.x;
  float f = beta[0]*o[(size_t)n*HID + h] + beta[1]*o[(size_t)(NN + n)*HID + h];
  red[0][h] = f*Wl[h*3+0]; red[1][h] = f*Wl[h*3+1]; red[2][h] = f*Wl[h*3+2];
  __syncthreads();
  for (int s = 64; s > 0; s >>= 1) {
    if (h < s) { red[0][h]+=red[0][h+s]; red[1][h]+=red[1][h+s]; red[2][h]+=red[2][h+s]; }
    __syncthreads();
  }
  if (h < 3) out[(size_t)n*3 + h] = red[h][0] + bl[h];
}

// ============================ host launcher ==================================
extern "C" void kernel_launch(void* const* d_in, const int* in_sizes, int n_in,
                              void* d_out, int out_size, void* d_ws, size_t ws_size,
                              hipStream_t stream) {
  const float* x      = (const float*)d_in[0];
  const int*   ei0    = (const int*)  d_in[1];
  const int*   ei1    = (const int*)  d_in[2];
  const float* Wproj  = (const float*)d_in[3];
  const float* bproj  = (const float*)d_in[4];
  const float* atts0  = (const float*)d_in[5];
  const float* attd0  = (const float*)d_in[6];
  const float* atts1  = (const float*)d_in[7];
  const float* attd1  = (const float*)d_in[8];
  const float* Wk     = (const float*)d_in[9];
  const float* bk     = (const float*)d_in[10];
  const float* q      = (const float*)d_in[11];
  const float* Wlin   = (const float*)d_in[12];
  const float* blin   = (const float*)d_in[13];
  float* out = (float*)d_out;

  // workspace carve-up (256B aligned)
  char* ws = (char*)d_ws; size_t off = 0;
  auto alloc = [&](size_t bytes)->char* {
    char* p = ws + off; off = (off + bytes + 255) & ~(size_t)255; return p;
  };
  _Float16* xh     = (_Float16*)alloc((size_t)NN*FIN*2);      // 51.2 MB
  _Float16* wpf    = (_Float16*)alloc(8*16*32*16*2);          // 128 KB
  _Float16* wkf    = (_Float16*)alloc(8*4*32*16*2);           // 32 KB
  float*    xp     = (float*)   alloc((size_t)NN*HID*4);      // 25.6 MB
  float*    as0    = (float*)   alloc((size_t)NN*HEADS*4);
  float*    ad0    = (float*)   alloc((size_t)NN*HEADS*4);
  float*    as1    = (float*)   alloc((size_t)NN*HEADS*4);
  float*    ad1    = (float*)   alloc((size_t)NN*HEADS*4);
  int*      amax   = (int*)     alloc((size_t)2*NN*HEADS*4);  // [2][N][8]
  float*    denom  = (float*)   alloc((size_t)2*NN*HEADS*4);
  float*    o      = (float*)   alloc((size_t)2*NN*HID*4);    // 51.2 MB
  _Float16* o16    = (_Float16*)alloc((size_t)2*NN*HID*2);    // 25.6 MB
  float*    colsum = (float*)   alloc(256*4);
  float*    beta   = (float*)   alloc(256);

  // 0) init accumulators
  init_ws<<<2048, 256, 0, stream>>>(o, denom, amax, colsum);
  // 1) x -> f16 ; W pre-swizzle to WMMA fragment order
  cvt_x<<<(NN*FIN/8 + 255)/256, 256, 0, stream>>>(x, xh);
  swizzle_w<<<(8*16*32*16 + 8*4*32*16 + 255)/256, 256, 0, stream>>>(Wproj, Wk, wpf, wkf);
  // 2) projection GEMM (WMMA)
  gemm_proj<<<NN/16, 256, 0, stream>>>(xh, wpf, bproj, xp);
  // 3) per-node attention scores
  node_scores<<<(NN*HEADS + 255)/256, 256, 0, stream>>>(xp, atts0, attd0, atts1, attd1,
                                                        as0, ad0, as1, ad1);
  // 4) edge softmax phases, metapath 0 then 1 (L2-resident gathers/atomics)
  edge_max<<<(EE+255)/256, 256, 0, stream>>>(ei0, as0, ad0, amax);
  edge_max<<<(EE+255)/256, 256, 0, stream>>>(ei1, as1, ad1, amax + (size_t)NN*HEADS);
  edge_den<<<(EE+255)/256, 256, 0, stream>>>(ei0, as0, ad0, amax, denom);
  edge_den<<<(EE+255)/256, 256, 0, stream>>>(ei1, as1, ad1, amax + (size_t)NN*HEADS,
                                             denom + (size_t)NN*HEADS);
  edge_msg<<<EE/8, 256, 0, stream>>>(ei0, as0, ad0, amax, denom, xp, o);
  edge_msg<<<EE/8, 256, 0, stream>>>(ei1, as1, ad1, amax + (size_t)NN*HEADS,
                                     denom + (size_t)NN*HEADS, xp, o + (size_t)NN*HID);
  // 5) relu + f16 copy
  relu_cvt<<<4096, 256, 0, stream>>>(o, o16);
  // 6) semantic attention: WMMA GEMM + column reduction, then beta softmax
  gemm_sem<<<dim3(NN/16, 2), 256, 0, stream>>>(o16, wkf, bk, colsum);
  sem_beta<<<1, 128, 0, stream>>>(colsum, q, beta);
  // 7) fused output + final linear
  final_lin<<<NN, 128, 0, stream>>>(o, beta, Wlin, blin, out);
}
